// Annalisa_29051158790789
// MI455X (gfx1250) — compile-verified, matrix operand order checked by reading, not run on hardware
//
#include <hip/hip_runtime.h>

typedef __attribute__((ext_vector_type(16))) _Float16     v16h;
typedef __attribute__((ext_vector_type(8)))  float        v8f;
typedef __attribute__((ext_vector_type(8)))  unsigned int v8u;

#define THRESH      20.0f
#define FT          51200                 // 80*640 elements per batch slice
#define BATCH       64
#define NAUX        32
#define NS          (BATCH * FT)          // 3,276,800 elements per channel
#define NWORDS      (NS / 32)             // 102,400 bit-words for strain mask
#define WORDS_PER_B (FT / 32)             // 1,600 words per batch slice

// ---------------------------------------------------------------------------
// Kernel 1: bit-pack strain peak mask. One u32 word (32 flat elements) per
// thread; peak(i) = q[i]>=20 && q[i]>q[i-1] && q[i]>q[i+1] on the fully
// flattened [B,F,T] signal (clamp provably redundant since 20 > 0).
// ---------------------------------------------------------------------------
__global__ __launch_bounds__(256) void strain_peak_bits(
    const float* __restrict__ qs, unsigned int* __restrict__ sb) {
  int w    = blockIdx.x * 256 + threadIdx.x;   // word id < NWORDS
  int base = w * 32;
  float e[34];
  const float4* p4 = reinterpret_cast<const float4*>(qs + base);
#pragma unroll
  for (int k = 0; k < 8; ++k) {
    float4 t = p4[k];
    e[1 + 4*k] = t.x; e[2 + 4*k] = t.y; e[3 + 4*k] = t.z; e[4 + 4*k] = t.w;
  }
  float pv = qs[(base == 0) ? 0 : base - 1];     // safe clamped address
  if (base == 0) pv = __builtin_inff();          // exclude flat index 0
  e[0] = pv;
  int ni = base + 32;
  float nx = qs[(ni >= NS) ? (NS - 1) : ni];
  if (ni >= NS) nx = __builtin_inff();           // exclude flat index NS-1
  e[33] = nx;
  unsigned int bits = 0u;
#pragma unroll
  for (int j = 0; j < 32; ++j) {
    float x = e[j + 1];
    int p = (x >= THRESH) & (x > e[j]) & (x > e[j + 2]);
    bits |= ((unsigned int)p) << j;
  }
  sb[w] = bits;
}

// ---------------------------------------------------------------------------
// Kernel 2: one block per (batch b, aux a). Streams the 200KB chunk with
// b128 loads, computes aux peak bits, ANDs with LDS-cached strain bits, and
// accumulates both populations with v_wmma_f32_16x16x32_f16:
//   A: per-lane f16 0/1 bits; halves h with h%8<4  = intersection bits,
//      halves with h%8>=4 = aux-peak bits  (K == h mod 8 in the ISA layout)
//   bP selects K%8<4 for every column, bQ selects K%8>=4.
//   Every D element equals its row-population sum -> sum(all D)/16 is the
//   count, independent of the C/D register layout.
// ---------------------------------------------------------------------------
__global__ __launch_bounds__(256) void aux_iou(
    const float* __restrict__ qa, const unsigned int* __restrict__ sb,
    float* __restrict__ out) {
  const int bb  = blockIdx.x;   // batch
  const int aa  = blockIdx.y;   // aux channel
  const int tid = threadIdx.x;

  __shared__ unsigned int ssb[WORDS_PER_B];
  __shared__ int   sc1;
  __shared__ float swA[8], swB[8];

  if (tid == 0) sc1 = 0;
  __syncthreads();

  // Stage this batch slice's strain bits into LDS; fold c1 popcount in.
  int myc1 = 0;
  for (int i = tid; i < WORDS_PER_B; i += 256) {
    unsigned int w = sb[bb * WORDS_PER_B + i];
    ssb[i] = w;
    myc1 += __popc(w);
  }
  atomicAdd(&sc1, myc1);

  const float* chan = qa + (size_t)aa * (size_t)NS;
  const unsigned int ONON = 0x3C003C00u;                 // two f16 1.0
  v8u bpU = {ONON, ONON, 0u, 0u, ONON, ONON, 0u, 0u};    // K%8 <  4
  v8u bqU = {0u, 0u, ONON, ONON, 0u, 0u, ONON, ONON};    // K%8 >= 4
  v16h bP = __builtin_bit_cast(v16h, bpU);
  v16h bQ = __builtin_bit_cast(v16h, bqU);
  v8f cA = {};   // accumulates 16 * inter
  v8f cB = {};   // accumulates 16 * c2
  v8u au = {};
  __syncthreads();

  const long gb = (long)bb * FT;
#pragma unroll 2
  for (int it = 0; it < 50; ++it) {
    int  base = it * 1024 + tid * 4;          // [0, FT)
    long gi   = gb + base;                    // flat index within channel
    const float4 v = *reinterpret_cast<const float4*>(chan + gi);
    float pv = chan[(gi == 0) ? 0 : gi - 1];
    if (gi == 0) pv = __builtin_inff();
    long nj = gi + 4;
    float nx = chan[(nj >= NS) ? (NS - 1) : nj];
    if (nj >= NS) nx = __builtin_inff();

    int p0 = (v.x >= THRESH) & (v.x > pv)  & (v.x > v.y);
    int p1 = (v.y >= THRESH) & (v.y > v.x) & (v.y > v.z);
    int p2 = (v.z >= THRESH) & (v.z > v.y) & (v.z > v.w);
    int p3 = (v.w >= THRESH) & (v.w > v.z) & (v.w > nx);

    unsigned int sw = ssb[base >> 5] >> (base & 31);
    int i0 = p0 & (int)(sw        & 1u);
    int i1 = p1 & (int)((sw >> 1) & 1u);
    int i2 = p2 & (int)((sw >> 2) & 1u);
    int i3 = p3 & (int)((sw >> 3) & 1u);

    int o = (it & 1) ? 4 : 0;                 // halves 0-7 / 8-15
    au[o + 0] = (i0 ? 0x3C00u : 0u) | (i1 ? 0x3C000000u : 0u);  // h%8 = 0,1
    au[o + 1] = (i2 ? 0x3C00u : 0u) | (i3 ? 0x3C000000u : 0u);  // h%8 = 2,3
    au[o + 2] = (p0 ? 0x3C00u : 0u) | (p1 ? 0x3C000000u : 0u);  // h%8 = 4,5
    au[o + 3] = (p2 ? 0x3C00u : 0u) | (p3 ? 0x3C000000u : 0u);  // h%8 = 6,7
    if (it & 1) {
      v16h a = __builtin_bit_cast(v16h, au);
      cA = __builtin_amdgcn_wmma_f32_16x16x32_f16(false, a, false, bP,
                                                  (short)0, cA, false, false);
      cB = __builtin_amdgcn_wmma_f32_16x16x32_f16(false, a, false, bQ,
                                                  (short)0, cB, false, false);
    }
  }

  // Layout-independent extraction: sum all 256 D elements per matrix.
  float tA = cA[0] + cA[1] + cA[2] + cA[3] + cA[4] + cA[5] + cA[6] + cA[7];
  float tB = cB[0] + cB[1] + cB[2] + cB[3] + cB[4] + cB[5] + cB[6] + cB[7];
#pragma unroll
  for (int off = 16; off >= 1; off >>= 1) {
    tA += __shfl_xor(tA, off, 32);
    tB += __shfl_xor(tB, off, 32);
  }
  int wave = tid >> 5;
  if ((tid & 31) == 0) { swA[wave] = tA; swB[wave] = tB; }
  __syncthreads();

  if (tid == 0) {
    float inter = 0.f, c2 = 0.f;
#pragma unroll
    for (int w = 0; w < 8; ++w) { inter += swA[w]; c2 += swB[w]; }
    inter *= 0.0625f;            // /16 columns, exact
    c2    *= 0.0625f;
    float c1  = (float)sc1;
    float uni = c1 + c2 - inter;
    float jac, ratio;
    if (uni == 0.f) { jac = 1.f; ratio = 1.f; }                 // zero_union
    else { jac = inter / uni; ratio = (c1 > 0.f) ? (inter / c1) : 0.f; }
    int idx = aa * BATCH + bb;
    out[idx] = jac;                       // iou,  [N_aux*B]
    out[NAUX * BATCH + idx] = ratio;      // corr, [N_aux*B]
  }
}

extern "C" void kernel_launch(void* const* d_in, const int* in_sizes, int n_in,
                              void* d_out, int out_size, void* d_ws, size_t ws_size,
                              hipStream_t stream) {
  const float* qs = (const float*)d_in[0];   // qt_strain [64,80,640]
  const float* qa = (const float*)d_in[1];   // qt_aux [32,64,80,640]
  float* out = (float*)d_out;                // 4096 f32: iou(2048) ++ corr(2048)
  unsigned int* sb = (unsigned int*)d_ws;    // 102,400 u32 strain bit-mask

  strain_peak_bits<<<NWORDS / 256, 256, 0, stream>>>(qs, sb);
  dim3 grid(BATCH, NAUX);
  aux_iou<<<grid, 256, 0, stream>>>(qa, sb, out);
}
